// SoftQuantize_13340168421721
// MI455X (gfx1250) — compile-verified, matrix operand order checked by reading, not run on hardware
//
#include <hip/hip_runtime.h>
#include <math.h>

typedef __attribute__((ext_vector_type(16))) __bf16 v16bf;
typedef __attribute__((ext_vector_type(8)))  __bf16 v8bf;
typedef __attribute__((ext_vector_type(8)))  float  v8f;
typedef __attribute__((ext_vector_type(4)))  float  v4f;

#define BSZ   32
#define CCH   64
#define HHH   128
#define HWW   128
#define DDIM  256
#define KKK   512
#define NTOK  131072              // 32*64*64 tokens
#define A_STRIDE 264              // 256 + 8 halves pad (keeps 16B alignment, kills bank conflicts)

// workspace layout (bytes)
#define WS_BHI   0u
#define WS_BLO   (256u*1024u)
#define WS_ECOL  (512u*1024u)
#define WS_CNORM (1024u*1024u)

// ---------------- pack kernels ----------------

// Pack E[d][k] (== embed, already row-major [256][512]) into WMMA-B swizzled bf16 hi/lo tiles.
// Tile (kt,nt): lane l holds column n = nt*16+(l&15); half j maps to K-offset c0 + (j<8 ? j : j+8),
// c0 = 8*(l>>4)  -- mirrors the ISA 16-bit operand layout.
__global__ void pack_b_kernel(const float* __restrict__ embed,
                              __bf16* __restrict__ bhi, __bf16* __restrict__ blo) {
    int gid = blockIdx.x * blockDim.x + threadIdx.x;    // 131072 total
    int j    = gid & 15;
    int l    = (gid >> 4) & 31;
    int tile = gid >> 9;                                // kt*32 + nt
    int nt = tile & 31;
    int kt = tile >> 5;
    int c0 = 8 * (l >> 4);
    int jj = (j < 8) ? j : (j + 8);
    int drow = kt * 32 + c0 + jj;
    int col  = nt * 16 + (l & 15);
    float f  = embed[drow * KKK + col];
    __bf16 h = (__bf16)f;
    __bf16 lo = (__bf16)(f - (float)h);
    bhi[gid] = h;
    blo[gid] = lo;
}

// Ecol[k*256 + d] = E[d][k]  (transposed f32 copy for the final gather)
__global__ void pack_ecol_kernel(const float* __restrict__ embed, float* __restrict__ ecol) {
    int gid = blockIdx.x * blockDim.x + threadIdx.x;    // 131072 total
    int k = gid >> 8;
    int d = gid & 255;
    ecol[gid] = embed[d * KKK + k];
}

// cnorm[k] = sum_d E[d][k]^2
__global__ void pack_cnorm_kernel(const float* __restrict__ embed, float* __restrict__ cn) {
    int k = blockIdx.x * blockDim.x + threadIdx.x;      // 512 total
    float s = 0.0f;
    for (int d = 0; d < DDIM; ++d) {
        float v = embed[d * KKK + k];
        s += v * v;
    }
    cn[k] = s;
}

// ---------------- main kernel ----------------
// grid = 2048 (one (b, rh) row = 64 tokens), block = 256 threads (8 waves, wave32).
__launch_bounds__(256)
__global__ void vq_main_kernel(const float*  __restrict__ x,
                               const __bf16* __restrict__ bhi,
                               const __bf16* __restrict__ blo,
                               const float*  __restrict__ ecol,
                               const float*  __restrict__ cnorm,
                               float* __restrict__ quant,
                               float* __restrict__ eind,
                               float* __restrict__ diffp,
                               float* __restrict__ perpp) {
    __shared__ __bf16 sAh[64 * A_STRIDE];
    __shared__ __bf16 sAl[64 * A_STRIDE];
    __shared__ unsigned long long skey[64];

    const int tid  = threadIdx.x;
    const int lane = tid & 31;
    const int wave = tid >> 5;
    const int blk  = blockIdx.x;
    const int b    = blk >> 6;
    const int rh   = blk & 63;

    if (tid < 64) skey[tid] = 0xFFFFFFFFFFFFFFFFull;

    // ---- stage A: 128 rows (s0,c) of 128 floats -> bf16 hi/lo planes in LDS ----
    for (int it = 0; it < 16; ++it) {
        int rr = wave * 16 + it;          // 0..127
        int c  = rr & 63;
        int s0 = rr >> 6;
        const float* src = x + (((size_t)(b * CCH + c) * HHH) + (size_t)(rh * 2 + s0)) * HWW
                             + (size_t)lane * 4;
        v4f v = *(const v4f*)src;
#pragma unroll
        for (int q = 0; q < 4; ++q) {
            int w  = lane * 4 + q;
            int rw = w >> 1;              // token in row
            int s1 = w & 1;
            int d  = s0 * 128 + s1 * 64 + c;
            float f = v[q];
            __bf16 h  = (__bf16)f;
            __bf16 lo = (__bf16)(f - (float)h);
            sAh[rw * A_STRIDE + d] = h;
            sAl[rw * A_STRIDE + d] = lo;
        }
    }
    __syncthreads();

    // ---- GEMM: 64 tokens x 512 codes, D=256 reduction, bf16x3 split ----
    v8f acc[4][4];
#pragma unroll
    for (int mt = 0; mt < 4; ++mt)
#pragma unroll
        for (int ni = 0; ni < 4; ++ni)
            acc[mt][ni] = (v8f)0.0f;

    const int mrow = lane & 15;
    const int c0   = 8 * (lane >> 4);
    const int ntb  = wave * 4;

#pragma unroll 1
    for (int kt = 0; kt < 8; ++kt) {
        // prefetch next kt's B lines into cache
        if (kt < 7) {
            size_t pf = ((size_t)((kt + 1) * 32 + ntb) * 32 + lane) * 16;
            __builtin_prefetch(&bhi[pf], 0, 1);
            __builtin_prefetch(&blo[pf], 0, 1);
        }
        v16bf ah[4], al[4];
#pragma unroll
        for (int mt = 0; mt < 4; ++mt) {
            int base = (mt * 16 + mrow) * A_STRIDE + kt * 32 + c0;
            v8bf h0 = *(const v8bf*)&sAh[base];
            v8bf h1 = *(const v8bf*)&sAh[base + 16];
            v8bf l0 = *(const v8bf*)&sAl[base];
            v8bf l1 = *(const v8bf*)&sAl[base + 16];
#pragma unroll
            for (int i = 0; i < 8; ++i) {
                ah[mt][i] = h0[i]; ah[mt][8 + i] = h1[i];
                al[mt][i] = l0[i]; al[mt][8 + i] = l1[i];
            }
        }
#pragma unroll
        for (int ni = 0; ni < 4; ++ni) {
            int nt = ntb + ni;
            size_t boff = ((size_t)(kt * 32 + nt) * 32 + lane) * 16;
            v8bf bh0 = *(const v8bf*)&bhi[boff];
            v8bf bh1 = *(const v8bf*)&bhi[boff + 8];
            v8bf bl0 = *(const v8bf*)&blo[boff];
            v8bf bl1 = *(const v8bf*)&blo[boff + 8];
            v16bf bh, bl;
#pragma unroll
            for (int i = 0; i < 8; ++i) {
                bh[i] = bh0[i]; bh[8 + i] = bh1[i];
                bl[i] = bl0[i]; bl[8 + i] = bl1[i];
            }
#pragma unroll
            for (int mt = 0; mt < 4; ++mt) {
                acc[mt][ni] = __builtin_amdgcn_wmma_f32_16x16x32_bf16(
                    false, ah[mt], false, bh, (short)0, acc[mt][ni], false, false);
                acc[mt][ni] = __builtin_amdgcn_wmma_f32_16x16x32_bf16(
                    false, ah[mt], false, bl, (short)0, acc[mt][ni], false, false);
                acc[mt][ni] = __builtin_amdgcn_wmma_f32_16x16x32_bf16(
                    false, al[mt], false, bh, (short)0, acc[mt][ni], false, false);
            }
        }
    }

    // ---- argmin via packed (dist-key<<32 | col) u64 atomicMin ----
    const int rgrp = 8 * (lane >> 4);
    float cn[4];
#pragma unroll
    for (int ni = 0; ni < 4; ++ni) cn[ni] = cnorm[(ntb + ni) * 16 + mrow];

#pragma unroll
    for (int mt = 0; mt < 4; ++mt) {
#pragma unroll
        for (int v = 0; v < 8; ++v) {
            unsigned long long best = 0xFFFFFFFFFFFFFFFFull;
#pragma unroll
            for (int ni = 0; ni < 4; ++ni) {
                float dist = cn[ni] - 2.0f * acc[mt][ni][v];
                unsigned ub  = __float_as_uint(dist);
                unsigned key = (ub & 0x80000000u) ? ~ub : (ub | 0x80000000u);
                unsigned col = (unsigned)((ntb + ni) * 16 + mrow);
                unsigned long long k64 = ((unsigned long long)key << 32) | (unsigned long long)col;
                best = (k64 < best) ? k64 : best;
            }
            int token = mt * 16 + rgrp + v;
            atomicMin(&skey[token], best);
        }
    }
    __syncthreads();

    // ---- gather chosen codes and write quantize in NCHW (coalesced float4 rows) ----
    for (int it = 0; it < 16; ++it) {
        int rr = wave * 16 + it;
        int c  = rr & 63;
        int s0 = rr >> 6;
        v4f o;
#pragma unroll
        for (int q = 0; q < 4; ++q) {
            int w  = lane * 4 + q;
            int rw = w >> 1;
            int s1 = w & 1;
            unsigned idx = (unsigned)(skey[rw] & 0xFFFFFFFFull);
            int d = s0 * 128 + s1 * 64 + c;
            o[q] = ecol[(size_t)idx * 256 + (size_t)d];
        }
        float* dst = quant + (((size_t)(b * CCH + c) * HHH) + (size_t)(rh * 2 + s0)) * HWW
                           + (size_t)lane * 4;
        *(v4f*)dst = o;
    }

    if (tid < 64) {
        eind[(size_t)blk * 64 + tid] = (float)(unsigned)(skey[tid] & 0xFFFFFFFFull);
    }
    if (blk == 0 && tid == 0) {
        diffp[0] = 0.0f;
        float p = 1.0f / 512.0f + 1e-10f;                    // avg_probs value (faithful mean-over-K)
        perpp[0] = expf(-((float)NTOK / 512.0f) * logf(p));  // -> +inf in f32, matches reference
    }
}

// ---------------- launcher ----------------
extern "C" void kernel_launch(void* const* d_in, const int* in_sizes, int n_in,
                              void* d_out, int out_size, void* d_ws, size_t ws_size,
                              hipStream_t stream) {
    const float* x     = (const float*)d_in[0];   // [32,64,128,128]
    const float* embed = (const float*)d_in[1];   // [2,2,64,512] == E[256][512] row-major
    float* out = (float*)d_out;
    char*  ws  = (char*)d_ws;

    __bf16* bhi  = (__bf16*)(ws + WS_BHI);
    __bf16* blo  = (__bf16*)(ws + WS_BLO);
    float*  ecol = (float*)(ws + WS_ECOL);
    float*  cn   = (float*)(ws + WS_CNORM);

    pack_b_kernel    <<<512, 256, 0, stream>>>(embed, bhi, blo);
    pack_ecol_kernel <<<512, 256, 0, stream>>>(embed, ecol);
    pack_cnorm_kernel<<<2,   256, 0, stream>>>(embed, cn);

    float* quant = out;                                   // 33554432 floats
    float* diffp = out + (size_t)33554432;                // 1
    float* eind  = out + (size_t)33554433;                // 131072
    float* perpp = out + (size_t)33554433 + (size_t)131072; // 1

    vq_main_kernel<<<2048, 256, 0, stream>>>(x, bhi, blo, ecol, cn, quant, eind, diffp, perpp);
}
